// HSIFusionNetV25LightningPro_21268678049969
// MI455X (gfx1250) — matrix operand order repaired and checked by reading, not compiled, not run on hardware
//
#include <hip/hip_runtime.h>
#include <hip/hip_bf16.h>

typedef __attribute__((ext_vector_type(16))) __bf16 v16bf;
typedef __attribute__((ext_vector_type(8)))  float  v8f;
typedef __attribute__((ext_vector_type(4)))  int    v4i;

#define AS1 __attribute__((address_space(1)))
#define AS3 __attribute__((address_space(3)))

#if defined(__HIP_DEVICE_COMPILE__) && \
    __has_builtin(__builtin_amdgcn_global_load_async_to_lds_b128) && \
    __has_builtin(__builtin_amdgcn_s_wait_asynccnt)
#define USE_ASYNC_LDS 1
#else
#define USE_ASYNC_LDS 0
#endif

constexpr int kB   = 2;
constexpr int kN   = 2048;
constexpr int kC   = 1024;
constexpr int kH   = 8;
constexpr int kHD  = 128;
constexpr int kBH  = kB * kH;    // 16
constexpr int kM   = kB * kN;    // 4096
constexpr int k3C  = 3 * kC;     // 3072
constexpr int kKeep = kN / 2;    // 1024

// ---------------- device helpers ----------------

__device__ inline v8f vzero() {
  v8f z;
#pragma unroll
  for (int i = 0; i < 8; ++i) z[i] = 0.0f;
  return z;
}

// A-fragment (16x32 bf16, MxK): lane holds row M=lane%16, hs=lane/16 selects K-half.
// VGPR v, half j -> K = 16*hs + 2*v + j   (contiguous, matches B)
__device__ inline v16bf load_a_frag_bf16(const __bf16* row /*32 contiguous K halves*/, int hs) {
  v16bf a;
  unsigned* au = reinterpret_cast<unsigned*>(&a);
#pragma unroll
  for (int v = 0; v < 8; ++v) {
    int k = (hs << 4) + (v << 1);
    au[v] = *reinterpret_cast<const unsigned*>(row + k);
  }
  return a;
}

// Build an A-fragment from an f32 row (softmax probabilities), converting to bf16.
__device__ inline v16bf make_a_frag_f32(const float* row, int k0, int hs) {
  v16bf a;
  __bf16* ae = reinterpret_cast<__bf16*>(&a);
#pragma unroll
  for (int v = 0; v < 8; ++v) {
    int k = k0 + (hs << 4) + (v << 1);
    ae[2 * v]     = (__bf16)row[k];
    ae[2 * v + 1] = (__bf16)row[k + 1];
  }
  return a;
}

// Order-preserving f32 <-> u32 mapping for exact top-k threshold bisection.
__device__ inline unsigned mono_map(float f) {
  unsigned u = __float_as_uint(f);
  return (u & 0x80000000u) ? ~u : (u | 0x80000000u);
}
__device__ inline float mono_unmap(unsigned u) {
  return __uint_as_float((u & 0x80000000u) ? (u & 0x7FFFFFFFu) : ~u);
}

__device__ inline int wred_sum_i(int v) {
#pragma unroll
  for (int i = 16; i > 0; i >>= 1) v += __shfl_xor(v, i, 32);
  return v;
}
__device__ inline float wred_max_f(float v) {
#pragma unroll
  for (int i = 16; i > 0; i >>= 1) v = fmaxf(v, __shfl_xor(v, i, 32));
  return v;
}
__device__ inline float wred_sum_f(float v) {
#pragma unroll
  for (int i = 16; i > 0; i >>= 1) v += __shfl_xor(v, i, 32);
  return v;
}

// 16-byte copy global -> LDS: async (ASYNCcnt, bypasses VGPRs) or sync fallback.
__device__ inline void copy16_g2l(const __bf16* g, __bf16* l) {
#if USE_ASYNC_LDS
  __builtin_amdgcn_global_load_async_to_lds_b128((AS1 v4i*)g, (AS3 v4i*)l, 0, 0);
#else
  *reinterpret_cast<uint4*>(l) = *reinterpret_cast<const uint4*>(g);
#endif
}

// ---------------- K1: fp32 -> bf16 packing ----------------

__global__ void cvt_bf16_kernel(const float* __restrict__ x,
                                const float* __restrict__ qw,
                                const float* __restrict__ pw,
                                __bf16* __restrict__ xb,
                                __bf16* __restrict__ wb,
                                __bf16* __restrict__ pb) {
  const int SX = kM * kC;
  const int SW = k3C * kC;
  const int SP = kC * kC;
  const int total = SX + SW + SP;
  for (int i = blockIdx.x * blockDim.x + threadIdx.x; i < total;
       i += gridDim.x * blockDim.x) {
    if (i < SX)           xb[i]            = (__bf16)x[i];
    else if (i < SX + SW) wb[i - SX]       = (__bf16)qw[i - SX];
    else                  pb[i - SX - SW]  = (__bf16)pw[i - SX - SW];
  }
}

// ---------------- shared GEMM core: C(128x64) = A(MxK) * B^T, B row-major [n][k] ----
// Double-buffered: tile s+1 streams into the other LDS buffer (async-to-LDS)
// while tile s feeds the WMMAs. 3 async b128 chunks per thread per tile.

__device__ inline void stage_tile(const __bf16* __restrict__ A,
                                  const __bf16* __restrict__ Bm,
                                  int K, int m0, int n0, int k0,
                                  __bf16* sA, __bf16* sB, int tid) {
  // A tile: 128 rows x 32 halves = 8 KB = 512 x 16B chunks, 2 per thread
#pragma unroll
  for (int i = 0; i < 2; ++i) {
    int c   = tid + (i << 8);
    int r   = c >> 2;
    int off = (c & 3) << 3;  // halves
    copy16_g2l(A + (size_t)(m0 + r) * K + k0 + off, sA + r * 32 + off);
  }
  // B tile: 64 rows x 32 halves = 4 KB = 256 x 16B chunks, 1 per thread
  {
    int r   = tid >> 2;
    int off = (tid & 3) << 3;
    copy16_g2l(Bm + (size_t)(n0 + r) * K + k0 + off, sB + r * 32 + off);
  }
}

__device__ inline void gemm_core_bf16(const __bf16* __restrict__ A,
                                      const __bf16* __restrict__ Bm,
                                      int K, int m0, int n0,
                                      __bf16* sA0, __bf16* sB0,
                                      __bf16* sA1, __bf16* sB1,
                                      v8f c[4]) {
  const int tid  = threadIdx.x;
  const int wave = tid >> 5;
  const int lane = tid & 31;
  const int hs   = lane >> 4;
  const int lm   = lane & 15;
#pragma unroll
  for (int i = 0; i < 4; ++i) c[i] = vzero();

  const int S = K / 32;
  stage_tile(A, Bm, K, m0, n0, 0, sA0, sB0, tid);

  for (int s = 0; s < S; ++s) {
    __bf16* curA = (s & 1) ? sA1 : sA0;
    __bf16* curB = (s & 1) ? sB1 : sB0;
    __bf16* nxtA = (s & 1) ? sA0 : sA1;
    __bf16* nxtB = (s & 1) ? sB0 : sB1;
    if (s + 1 < S) {
      stage_tile(A, Bm, K, m0, n0, (s + 1) * 32, nxtA, nxtB, tid);
#if USE_ASYNC_LDS
      __builtin_amdgcn_s_wait_asynccnt(3);  // tile s done; 3 chunks of s+1 in flight
#endif
    } else {
#if USE_ASYNC_LDS
      __builtin_amdgcn_s_wait_asynccnt(0);
#endif
    }
    __syncthreads();

    v16bf a = load_a_frag_bf16(curA + (wave * 16 + lm) * 32, hs);
#pragma unroll
    for (int nt = 0; nt < 4; ++nt) {
      v16bf b = *reinterpret_cast<const v16bf*>(curB + (nt * 16 + lm) * 32 + (hs << 4));
      c[nt] = __builtin_amdgcn_wmma_f32_16x16x32_bf16(false, a, false, b,
                                                      (short)0, c[nt], false, false);
    }
    __syncthreads();  // done reading cur before it is restaged at s+2
  }
}

// ---------------- K2: QKV projection, scatter into per-head layouts ----------------

__global__ void qkv_gemm_kernel(const __bf16* __restrict__ xb,
                                const __bf16* __restrict__ wb,
                                const float* __restrict__ qkv_b,
                                __bf16* __restrict__ qkvb,  // [3][16][2048][128]
                                __bf16* __restrict__ vT) {  // [16][128][2048]
  __shared__ __align__(64) __bf16 sA0[128 * 32];
  __shared__ __align__(64) __bf16 sB0[64 * 32];
  __shared__ __align__(64) __bf16 sA1[128 * 32];
  __shared__ __align__(64) __bf16 sB1[64 * 32];
  const int n0 = blockIdx.x * 64;
  const int m0 = blockIdx.y * 128;
  v8f c[4];
  gemm_core_bf16(xb, wb, kC, m0, n0, sA0, sB0, sA1, sB1, c);

  const int tid  = threadIdx.x;
  const int wave = tid >> 5;
  const int lane = tid & 31;
  const int hs   = lane >> 4;
  const int lm   = lane & 15;
#pragma unroll
  for (int nt = 0; nt < 4; ++nt) {
#pragma unroll
    for (int r = 0; r < 8; ++r) {
      int m = m0 + wave * 16 + r + (hs << 3);
      int n = n0 + nt * 16 + lm;
      float val = c[nt][r] + qkv_b[n];
      int i3 = n >> 10;
      int h  = (n >> 7) & 7;
      int d  = n & 127;
      int b  = m >> 11;
      int t  = m & 2047;
      int g  = (b << 3) + h;
      __bf16 bv = (__bf16)val;
      qkvb[(((size_t)i3 * kBH + g) * kN + t) * kHD + d] = bv;
      if (i3 == 2) vT[((size_t)g * kHD + d) * kN + t] = bv;
    }
  }
}

// ---------------- K3: attention with exact top-k threshold + softmax ----------------

__global__ void attn_kernel(const __bf16* __restrict__ qb,
                            const __bf16* __restrict__ kb,
                            const __bf16* __restrict__ vT,
                            const float* __restrict__ temperature,
                            __bf16* __restrict__ attn_out) {
  __shared__ __align__(64) float  s_scores[16][kN];  // 128 KB (of 320 KB/WGP)
  __shared__ __align__(64) __bf16 s_q[16 * kHD];     // 4 KB

  const int g    = blockIdx.y;       // b*H + h
  const int row0 = blockIdx.x * 16;  // query row tile
  const int tid  = threadIdx.x;
  const int wave = tid >> 5;
  const int lane = tid & 31;
  const int hs   = lane >> 4;
  const int lm   = lane & 15;

  const __bf16* Q  = qb + (size_t)g * kN * kHD;
  const __bf16* Kk = kb + (size_t)g * kN * kHD;
  const __bf16* Vt = vT + (size_t)g * kHD * kN;

  const float temp   = fmaxf(temperature[0], 0.01f);
  const float sscale = 0.08838834764831845f / temp;  // 128^-0.5 / clip(temp)

  {  // stage q tile (16 x 128): 2048 halves, 8 halves (16B) per thread
    int o = tid * 8;
    *reinterpret_cast<uint4*>(s_q + o) =
        *reinterpret_cast<const uint4*>(Q + (size_t)(row0 + (o >> 7)) * kHD + (o & 127));
  }
  __syncthreads();

  v16bf aq[4];
#pragma unroll
  for (int kk = 0; kk < 4; ++kk)
    aq[kk] = load_a_frag_bf16(s_q + lm * kHD + kk * 32, hs);

  // scores tile row (16 x 2048): each wave owns every 8th key tile
  for (int jt = wave; jt < kN / 16; jt += 8) {
    if (jt + 8 < kN / 16)  // prefetch next key tile row for this lane
      __builtin_prefetch(Kk + (size_t)((jt + 8) * 16 + lm) * kHD, 0, 1);
    v8f c = vzero();
#pragma unroll
    for (int kk = 0; kk < 4; ++kk) {
      v16bf b = *reinterpret_cast<const v16bf*>(
          Kk + (size_t)(jt * 16 + lm) * kHD + kk * 32 + (hs << 4));
      c = __builtin_amdgcn_wmma_f32_16x16x32_bf16(false, aq[kk], false, b,
                                                  (short)0, c, false, false);
    }
#pragma unroll
    for (int r = 0; r < 8; ++r)
      s_scores[r + (hs << 3)][jt * 16 + lm] = c[r] * sscale;
  }
  __syncthreads();

  // exact k-th largest threshold via 32-step bisection on monotone u32 space, then softmax
#pragma unroll
  for (int rr = 0; rr < 2; ++rr) {
    const int row = wave * 2 + rr;
    unsigned lo = 0u, hi = 0xFFFFFFFFu;
    while (lo < hi) {
      unsigned mid = lo + ((hi - lo) >> 1) + 1u;  // upper mid, no overflow
      int cnt = 0;
      for (int j = lane; j < kN; j += 32)
        cnt += (mono_map(s_scores[row][j]) >= mid) ? 1 : 0;
      cnt = wred_sum_i(cnt);
      if (cnt >= kKeep) lo = mid; else hi = mid - 1u;
    }
    const float thresh = mono_unmap(lo);

    float mx = -3.4e38f;
    for (int j = lane; j < kN; j += 32) {
      float sv = s_scores[row][j];
      if (sv >= thresh) mx = fmaxf(mx, sv);
    }
    mx = wred_max_f(mx);

    float sm = 0.0f;
    for (int j = lane; j < kN; j += 32) {
      float sv = s_scores[row][j];
      sm += (sv >= thresh) ? __expf(sv - mx) : 0.0f;
    }
    sm = wred_sum_f(sm);
    const float inv = 1.0f / sm;

    for (int j = lane; j < kN; j += 32) {
      float sv = s_scores[row][j];
      s_scores[row][j] = (sv >= thresh) ? __expf(sv - mx) * inv : 0.0f;
    }
  }
  __syncthreads();

  // out(16 x 128) = P(16 x 2048) @ V(2048 x 128); each wave owns 16 d-columns
  v8f co = vzero();
  for (int k0 = 0; k0 < kN; k0 += 32) {
    if (k0 + 32 < kN)
      __builtin_prefetch(Vt + (size_t)(wave * 16 + lm) * kN + k0 + 32, 0, 1);
    v16bf a = make_a_frag_f32(&s_scores[lm][0], k0, hs);
    v16bf b = *reinterpret_cast<const v16bf*>(
        Vt + (size_t)(wave * 16 + lm) * kN + k0 + (hs << 4));
    co = __builtin_amdgcn_wmma_f32_16x16x32_bf16(false, a, false, b,
                                                 (short)0, co, false, false);
  }
  const int b_ = g >> 3;
  const int h_ = g & 7;
#pragma unroll
  for (int r = 0; r < 8; ++r) {
    int t = row0 + r + (hs << 3);
    int d = wave * 16 + lm;
    attn_out[((size_t)(b_ * kN + t)) * kC + h_ * kHD + d] = (__bf16)co[r];
  }
}

// ---------------- K4: output projection (f32 epilogue into d_out) ----------------

__global__ void proj_gemm_kernel(const __bf16* __restrict__ ab,
                                 const __bf16* __restrict__ pw,
                                 const float* __restrict__ proj_b,
                                 float* __restrict__ out) {
  __shared__ __align__(64) __bf16 sA0[128 * 32];
  __shared__ __align__(64) __bf16 sB0[64 * 32];
  __shared__ __align__(64) __bf16 sA1[128 * 32];
  __shared__ __align__(64) __bf16 sB1[64 * 32];
  const int n0 = blockIdx.x * 64;
  const int m0 = blockIdx.y * 128;
  v8f c[4];
  gemm_core_bf16(ab, pw, kC, m0, n0, sA0, sB0, sA1, sB1, c);

  const int tid  = threadIdx.x;
  const int wave = tid >> 5;
  const int lane = tid & 31;
  const int hs   = lane >> 4;
  const int lm   = lane & 15;
#pragma unroll
  for (int nt = 0; nt < 4; ++nt) {
#pragma unroll
    for (int r = 0; r < 8; ++r) {
      int m = m0 + wave * 16 + r + (hs << 3);
      int n = n0 + nt * 16 + lm;
      out[(size_t)m * kC + n] = c[nt][r] + proj_b[n];
    }
  }
}

// ---------------- host launcher ----------------

extern "C" void kernel_launch(void* const* d_in, const int* in_sizes, int n_in,
                              void* d_out, int out_size, void* d_ws, size_t ws_size,
                              hipStream_t stream) {
  (void)in_sizes; (void)n_in; (void)out_size; (void)ws_size;
  const float* x           = (const float*)d_in[0];
  const float* qkv_w       = (const float*)d_in[1];
  const float* qkv_b       = (const float*)d_in[2];
  const float* proj_w      = (const float*)d_in[3];
  const float* proj_b      = (const float*)d_in[4];
  const float* temperature = (const float*)d_in[5];
  float* out = (float*)d_out;

  char* ws = (char*)d_ws;
  size_t off = 0;
  __bf16* xb   = (__bf16*)(ws + off); off += (size_t)kM  * kC * 2;               // 8 MB
  __bf16* wb   = (__bf16*)(ws + off); off += (size_t)k3C * kC * 2;               // 6 MB
  __bf16* pb   = (__bf16*)(ws + off); off += (size_t)kC  * kC * 2;               // 2 MB
  __bf16* qkvb = (__bf16*)(ws + off); off += (size_t)3 * kBH * kN * kHD * 2;     // 24 MB
  __bf16* vT   = (__bf16*)(ws + off); off += (size_t)kBH * kHD * kN * 2;         // 8 MB
  __bf16* ao   = (__bf16*)(ws + off); off += (size_t)kM * kC * 2;                // 8 MB

  cvt_bf16_kernel<<<4096, 256, 0, stream>>>(x, qkv_w, proj_w, xb, wb, pb);

  qkv_gemm_kernel<<<dim3(k3C / 64, kM / 128), 256, 0, stream>>>(xb, wb, qkv_b, qkvb, vT);

  const __bf16* qq = qkvb;
  const __bf16* kk = qkvb + (size_t)kBH * kN * kHD;
  attn_kernel<<<dim3(kN / 16, kBH), 256, 0, stream>>>(qq, kk, vT, temperature, ao);

  proj_gemm_kernel<<<dim3(kC / 64, kM / 128), 256, 0, stream>>>(ao, pb, proj_b, out);
}
